// AttentionPerformer_15401752723816
// MI455X (gfx1250) — matrix operand order repaired and checked by reading, not compiled, not run on hardware
//
#include <hip/hip_runtime.h>
#include <hip/hip_bf16.h>

// ---------------------------------------------------------------------------
// Performer attention for MI455X (gfx1250, wave32, WMMA).
// Shapes: B=8 N=4096 C=768 H=12 HD=64 M=32.
// Memory-bound (155 GFLOP vs ~1GB min traffic @23.3TB/s): convert once to
// f16, run GEMMs on v_wmma_f32_16x16x32_f16 (f32 accum); QP@kptv^T uses
// v_wmma_f32_16x16x4_f32 since kptv sums 4096 exp() terms (f16 range risk).
// LDS staging in the per-head stage uses GLOBAL_LOAD_ASYNC_TO_LDS_B128
// (ASYNCcnt) when the toolchain exposes the builtin.
// ---------------------------------------------------------------------------

typedef __attribute__((ext_vector_type(16))) _Float16 v16h;
typedef __attribute__((ext_vector_type(8)))  _Float16 v8h;
typedef __attribute__((ext_vector_type(4)))  _Float16 v4h;
typedef __attribute__((ext_vector_type(8)))  float    v8f;
typedef __attribute__((ext_vector_type(2)))  float    v2f;
typedef __attribute__((ext_vector_type(4)))  int      v4i;

#define BDIM   8
#define NSEQ   4096
#define CDIM   768
#define HEADS  12
#define HDIM   64
#define MFEAT  32
#define ROWS   (BDIM * NSEQ)       // 32768
#define C3     (3 * CDIM)          // 2304

#if defined(__has_builtin)
#if __has_builtin(__builtin_amdgcn_global_load_async_to_lds_b128)
#define HAVE_ASYNC_LDS 1
#endif
#endif

#ifdef HAVE_ASYNC_LDS
// Builtin signature (from hipcc diagnostic): first param is
// 'v4i addrspace(1)*' -> (global v4i* src, shared v4i* dst, imm offset, imm cpol)
typedef __attribute__((address_space(1))) v4i* gvec_t;
typedef __attribute__((address_space(3))) v4i* lvec_t;
__device__ __forceinline__ void async_ld_b128(const void* gsrc, void* ldst) {
  __builtin_amdgcn_global_load_async_to_lds_b128((gvec_t)gsrc, (lvec_t)ldst, 0, 0);
}
__device__ __forceinline__ void async_wait0() {
#if __has_builtin(__builtin_amdgcn_s_wait_asynccnt)
  __builtin_amdgcn_s_wait_asynccnt(0);
#else
  asm volatile("s_wait_asynccnt 0" ::: "memory");
#endif
}
#endif

__device__ __forceinline__ v8f wmma_f16(v16h a, v16h b, v8f c) {
  return __builtin_amdgcn_wmma_f32_16x16x32_f16(false, a, false, b,
                                                (short)0, c, false, false);
}
__device__ __forceinline__ v8f wmma_f32(v2f a, v2f b, v8f c) {
  return __builtin_amdgcn_wmma_f32_16x16x4_f32(false, a, false, b,
                                               (short)0, c, false, false);
}

// A-fragment: 16x32 f16, rows of row-major A[row0.., k0..k0+31], stride lda.
// ISA layout: lanes 0-15 row=lane, K in {8hf..8hf+7} (v0-3) and {16+8hf..} (v4-7).
__device__ __forceinline__ v16h load_a_frag(const _Float16* __restrict__ A,
                                            int lda, int row0, int k0, int lane) {
  int r  = row0 + (lane & 15);
  int hf = lane >> 4;
  const _Float16* p = A + (size_t)r * lda + k0 + 8 * hf;
  v8h lo = *(const v8h*)p;
  v8h hi = *(const v8h*)(p + 16);
  v16h a;
#pragma unroll
  for (int i = 0; i < 8; ++i) { a[i] = lo[i]; a[8 + i] = hi[i]; }
  return a;
}

// B-fragment: 32x16 f16 representing W^T tile. W row-major [Ncols x K]:
// B[k][n] = W[col0+n][k0+k]. Lane n=lane%16 holds column n, K in {8hf..8hf+7}
// (elements 0-7) and {16+8hf..} (elements 8-15), matching the A-fragment.
__device__ __forceinline__ v16h load_bT_frag(const _Float16* __restrict__ W,
                                             int ldw, int k0, int col0, int lane) {
  int n  = col0 + (lane & 15);
  int kh = (lane >> 4) * 8;
  const _Float16* p = W + (size_t)n * ldw + k0 + kh;
  v8h lo = *(const v8h*)p;
  v8h hi = *(const v8h*)(p + 16);
  v16h b;
#pragma unroll
  for (int i = 0; i < 8; ++i) { b[i] = lo[i]; b[8 + i] = hi[i]; }
  return b;
}

// ---------------------------------------------------------------------------
// Kernel 0: f32 -> f16 conversion (grid-stride)
// ---------------------------------------------------------------------------
__global__ void cvt_f32_to_f16(const float* __restrict__ src,
                               _Float16* __restrict__ dst, size_t n) {
  size_t i = (size_t)blockIdx.x * blockDim.x + threadIdx.x;
  size_t stride = (size_t)gridDim.x * blockDim.x;
  for (; i < n; i += stride) dst[i] = (_Float16)src[i];
}

// ---------------------------------------------------------------------------
// Kernel 1/5: C = A @ W^T + bias.  A [Nr x K] f16, W [Nc x K] f16,
// bias [Nc] f32, out [Nr x Nc] (f16 or f32).
// Block = 256 threads = 8 waves; wave tile = 16x64 (4 accum frags, A reused).
// Grid = (Nr/128, Nc/64). Next k-tile prefetched (global_prefetch_b8).
// ---------------------------------------------------------------------------
template <typename OutT>
__global__ void gemm_wmma(const _Float16* __restrict__ A,
                          const _Float16* __restrict__ W,
                          const float* __restrict__ bias,
                          OutT* __restrict__ Cout,
                          int Nr, int K, int Nc) {
  int lane = threadIdx.x & 31;
  int wave = threadIdx.x >> 5;
  int row0 = blockIdx.x * 128 + wave * 16;
  int col0 = blockIdx.y * 64;

  v8f c[4] = {};
  for (int k0 = 0; k0 < K; k0 += 32) {
    if (k0 + 32 < K) {  // prefetch next k-tile (one cacheline per lane)
      __builtin_prefetch(A + (size_t)(row0 + (lane & 15)) * K + k0 + 32, 0, 1);
      __builtin_prefetch(W + (size_t)(col0 + (lane & 31)) * K + k0 + 32, 0, 1);
    }
    v16h a = load_a_frag(A, K, row0, k0, lane);
#pragma unroll
    for (int j = 0; j < 4; ++j) {
      v16h b = load_bT_frag(W, K, k0, col0 + 16 * j, lane);
      c[j] = wmma_f16(a, b, c[j]);
    }
  }
  int n = lane & 15, hf = lane >> 4;
#pragma unroll
  for (int j = 0; j < 4; ++j) {
    int col = col0 + 16 * j + n;
    float bv = bias[col];
#pragma unroll
    for (int v = 0; v < 8; ++v) {
      int row = row0 + v + 8 * hf;
      float val = c[j][v] + bv;
      if constexpr (sizeof(OutT) == 2)
        Cout[(size_t)row * Nc + col] = (OutT)(_Float16)val;
      else
        Cout[(size_t)row * Nc + col] = (OutT)val;
    }
  }
}

// ---------------------------------------------------------------------------
// Kernel 2: prm_exp.  out[bh,t,m] = exp((X @ w_h^T)[t,m] - 0.5*||X[t]||^2)/sqrt(M)
// X = head slice of kqv (k or q selected by sSel). ||x||^2 computed from the
// same f16 A fragments via lane-sum + shfl_xor(16).
// Grid = (N/128, B*H), block = 256 (8 waves x 16 rows).
// ---------------------------------------------------------------------------
__global__ void prm_exp_kernel(const _Float16* __restrict__ kqvh,
                               const _Float16* __restrict__ wh,
                               _Float16* __restrict__ outp, int sSel) {
  int bh = blockIdx.y;
  int b = bh / HEADS, h = bh % HEADS;
  int lane = threadIdx.x & 31, wave = threadIdx.x >> 5;
  int t0 = blockIdx.x * 128 + wave * 16;

  const _Float16* Abase = kqvh + (size_t)(b * NSEQ) * C3 + sSel * CDIM + h * HDIM;
  const _Float16* Wh = wh + (size_t)h * MFEAT * HDIM;

  v8f c0 = {}, c1 = {};
  float ss = 0.f;
#pragma unroll
  for (int d0 = 0; d0 < HDIM; d0 += 32) {
    v16h a = load_a_frag(Abase, C3, t0, d0, lane);
#pragma unroll
    for (int i = 0; i < 16; ++i) { float f = (float)a[i]; ss += f * f; }
    v16h b0 = load_bT_frag(Wh, HDIM, d0, 0, lane);
    v16h b1 = load_bT_frag(Wh, HDIM, d0, 16, lane);
    c0 = wmma_f16(a, b0, c0);
    c1 = wmma_f16(a, b1, c1);
  }
  ss += __shfl_xor(ss, 16, 32);      // full row sum-of-squares on every lane
  float xd = 0.5f * ss;              // lane L holds xd of row (L & 15)

  const float scale = 0.17677669529663687f;  // 1/sqrt(32)
  int n = lane & 15, hf = lane >> 4;
  _Float16* orow = outp + ((size_t)bh * NSEQ + t0) * MFEAT;
#pragma unroll
  for (int v = 0; v < 8; ++v) {
    int r = v + 8 * hf;
    float xdr = __shfl(xd, r, 32);
    orow[(size_t)r * MFEAT + n]      = (_Float16)(__expf(c0[v] - xdr) * scale);
    orow[(size_t)r * MFEAT + 16 + n] = (_Float16)(__expf(c1[v] - xdr) * scale);
  }
}

// ---------------------------------------------------------------------------
// Kernel 3: kptvT[bh][m][d] = sum_t kp[bh,t,m] * v[b,h,t,d]   ([32 x 64], f32)
// One block per (b,h); 8 waves each own one 16x16 tile of the 32x64 result.
// KP and V chunks staged in LDS via async global->LDS b128 copies (ASYNCcnt)
// when available; transposed reads build the KP^T A-fragments.
// ---------------------------------------------------------------------------
__global__ void kptv_kernel(const _Float16* __restrict__ kqvh,
                            const _Float16* __restrict__ kp,
                            float* __restrict__ kptvT) {
  int bh = blockIdx.x;
  int b = bh / HEADS, h = bh % HEADS;
  __shared__ _Float16 kpT[32 * MFEAT];   // [t_local][m]
  __shared__ _Float16 vT[32 * HDIM];     // [t_local][d]

  int tid = threadIdx.x, lane = tid & 31, wave = tid >> 5;
  int m0 = (wave & 1) * 16;
  int d0 = (wave >> 1) * 16;

  const _Float16* kpbase = kp + (size_t)bh * NSEQ * MFEAT;
  const _Float16* vbase  = kqvh + (size_t)b * NSEQ * C3 + 2 * CDIM + h * HDIM;

  v8f acc = {};
  for (int t0 = 0; t0 < NSEQ; t0 += 32) {
#ifdef HAVE_ASYNC_LDS
    // KP chunk: 32x32 halves = 128 b128 transfers (threads 0..127)
    if (tid < 128) {
      int tl = tid >> 2, mm = (tid & 3) * 8;
      async_ld_b128(kpbase + (size_t)(t0 + tl) * MFEAT + mm, kpT + tid * 8);
    }
    // V chunk: 32x64 halves = 256 b128 transfers (all threads)
    {
      int tl = tid >> 3, dd = (tid & 7) * 8;
      async_ld_b128(vbase + (size_t)(t0 + tl) * C3 + dd, vT + tid * 8);
    }
    async_wait0();
#else
    {  // stage KP chunk: 32x32 halves, 4 per thread
      int idx = tid * 4;
      int tl = idx >> 5, mm = idx & 31;
      *(v4h*)(kpT + idx) = *(const v4h*)(kpbase + (size_t)(t0 + tl) * MFEAT + mm);
    }
    {  // stage V chunk: 32x64 halves, 8 per thread
      int idx = tid * 8;
      int tl = idx >> 6, dd = idx & 63;
      *(v8h*)(vT + idx) = *(const v8h*)(vbase + (size_t)(t0 + tl) * C3 + dd);
    }
#endif
    __syncthreads();

    v16h a;  // A = KP^T tile: A[m_row, k=t]
    {
      int m = m0 + (lane & 15);
      int hf = lane >> 4;
#pragma unroll
      for (int i = 0; i < 8; ++i) {
        a[i]     = kpT[(8 * hf + i) * MFEAT + m];
        a[8 + i] = kpT[(16 + 8 * hf + i) * MFEAT + m];
      }
    }
    v16h bf;  // B = V tile: B[k=t, n=d]
    {
      int d = d0 + (lane & 15);
      int kh = (lane >> 4) * 8;
#pragma unroll
      for (int i = 0; i < 16; ++i) bf[i] = vT[(kh + i + (i >= 8 ? 8 : 0)) * HDIM + d];
    }
    acc = wmma_f16(a, bf, acc);
    __syncthreads();
  }
  int n = lane & 15, hf = lane >> 4;
#pragma unroll
  for (int v = 0; v < 8; ++v) {
    int m = m0 + v + 8 * hf;
    kptvT[((size_t)bh * MFEAT + m) * HDIM + d0 + n] = acc[v];
  }
}

// ---------------------------------------------------------------------------
// Kernel 3b: ks[bh][m] = sum_t kp[bh,t,m]
// ---------------------------------------------------------------------------
__global__ void ks_kernel(const _Float16* __restrict__ kp, float* __restrict__ ks) {
  int bh = blockIdx.x;
  int m = threadIdx.x & 31, chunk = threadIdx.x >> 5;  // 8 chunks of 512 t
  const _Float16* base = kp + (size_t)bh * NSEQ * MFEAT;
  float s = 0.f;
  for (int t = chunk * 512; t < chunk * 512 + 512; ++t)
    s += (float)base[(size_t)t * MFEAT + m];
  __shared__ float red[8][32];
  red[chunk][m] = s;
  __syncthreads();
  if (threadIdx.x < 32) {
    float tot = 0.f;
#pragma unroll
    for (int i = 0; i < 8; ++i) tot += red[i][threadIdx.x];
    ks[bh * MFEAT + threadIdx.x] = tot;
  }
}

// ---------------------------------------------------------------------------
// Kernel 4: y[t,d] = (sum_m qp[t,m]*kptvT[m,d]) / (sum_m qp[t,m]*ks[m] + eps)
// f32 WMMA (16x16x4) since kptvT is f32. kptvT + ks staged in LDS.
// Output written as f16 into yh [ROWS x C] at column h*64+d for final GEMM.
// Grid = (N/128, B*H), block = 256.
// ---------------------------------------------------------------------------
__global__ void y_kernel(const _Float16* __restrict__ qp,
                         const float* __restrict__ kptvT,
                         const float* __restrict__ ks,
                         _Float16* __restrict__ yh) {
  int bh = blockIdx.y;
  int b = bh / HEADS, h = bh % HEADS;
  __shared__ float Cs[MFEAT * HDIM];   // [m][d]
  __shared__ float kss[MFEAT];
  int tid = threadIdx.x, lane = tid & 31, wave = tid >> 5;

  for (int i = tid; i < MFEAT * HDIM; i += 256)
    Cs[i] = kptvT[(size_t)bh * MFEAT * HDIM + i];
  if (tid < MFEAT) kss[tid] = ks[bh * MFEAT + tid];
  __syncthreads();

  int t0 = blockIdx.x * 128 + wave * 16;
  const _Float16* qbase = qp + ((size_t)bh * NSEQ + t0) * MFEAT;

  v8f c[4] = {};
#pragma unroll
  for (int mc = 0; mc < 8; ++mc) {       // K chunks of 4 over m
    v2f a;                               // A 16x4 f32: K = 2*hf + v
    {
      int r = lane & 15;
      int kb = mc * 4 + (lane >> 4) * 2;
      a.x = (float)qbase[(size_t)r * MFEAT + kb];
      a.y = (float)qbase[(size_t)r * MFEAT + kb + 1];
    }
#pragma unroll
    for (int dt = 0; dt < 4; ++dt) {     // B 4x16 f32: K = 2*hf + v
      v2f bf;
      int col = dt * 16 + (lane & 15);
      int kb = mc * 4 + (lane >> 4) * 2;
      bf.x = Cs[(kb + 0) * HDIM + col];
      bf.y = Cs[(kb + 1) * HDIM + col];
      c[dt] = wmma_f32(a, bf, c[dt]);
    }
  }
  // D[t] = qp[t,:] . ks  (lane L computes row L&15)
  float Dv = 0.f;
  {
    int r = lane & 15;
    const _Float16* q = qbase + (size_t)r * MFEAT;
#pragma unroll
    for (int mm = 0; mm < MFEAT; ++mm) Dv += (float)q[mm] * kss[mm];
  }
  int n = lane & 15, hf = lane >> 4;
  _Float16* obase = yh + (size_t)(b * NSEQ + t0) * CDIM + h * HDIM;
#pragma unroll
  for (int v = 0; v < 8; ++v) {
    int r = v + 8 * hf;
    float Dr = __shfl(Dv, r, 32);
    float inv = 1.0f / (Dr + 1e-8f);
#pragma unroll
    for (int dt = 0; dt < 4; ++dt)
      obase[(size_t)r * CDIM + dt * 16 + n] = (_Float16)(c[dt][v] * inv);
  }
}

// ---------------------------------------------------------------------------
// Launch: inputs = {x, kqv_w, kqv_b, proj_w, proj_b, w}
// ---------------------------------------------------------------------------
extern "C" void kernel_launch(void* const* d_in, const int* in_sizes, int n_in,
                              void* d_out, int out_size, void* d_ws, size_t ws_size,
                              hipStream_t stream) {
  const float* x      = (const float*)d_in[0];
  const float* kqv_w  = (const float*)d_in[1];
  const float* kqv_b  = (const float*)d_in[2];
  const float* proj_w = (const float*)d_in[3];
  const float* proj_b = (const float*)d_in[4];
  const float* wmat   = (const float*)d_in[5];
  float* out = (float*)d_out;

  // workspace carve-up (all 256B aligned)
  char* ws = (char*)d_ws;
  size_t off = 0;
  auto alloc = [&](size_t bytes) {
    void* p = ws + off;
    off += (bytes + 255) & ~(size_t)255;
    return p;
  };
  _Float16* xh     = (_Float16*)alloc((size_t)ROWS * CDIM * 2);      // 50.3 MB
  _Float16* wkqvh  = (_Float16*)alloc((size_t)C3 * CDIM * 2);        //  3.5 MB
  _Float16* projh  = (_Float16*)alloc((size_t)CDIM * CDIM * 2);      //  1.2 MB
  _Float16* wh     = (_Float16*)alloc((size_t)HEADS * MFEAT * HDIM * 2);
  _Float16* kqvh   = (_Float16*)alloc((size_t)ROWS * C3 * 2);        //  151 MB
  _Float16* kp     = (_Float16*)alloc((size_t)BDIM * HEADS * NSEQ * MFEAT * 2);
  _Float16* qp     = (_Float16*)alloc((size_t)BDIM * HEADS * NSEQ * MFEAT * 2);
  float*    kptvT  = (float*)alloc((size_t)BDIM * HEADS * MFEAT * HDIM * 4);
  float*    ksbuf  = (float*)alloc((size_t)BDIM * HEADS * MFEAT * 4);
  _Float16* yh     = (_Float16*)alloc((size_t)ROWS * CDIM * 2);      // 50.3 MB

  // 0) conversions to f16
  cvt_f32_to_f16<<<4096, 256, 0, stream>>>(x, xh, (size_t)ROWS * CDIM);
  cvt_f32_to_f16<<<2048, 256, 0, stream>>>(kqv_w, wkqvh, (size_t)C3 * CDIM);
  cvt_f32_to_f16<<<1024, 256, 0, stream>>>(proj_w, projh, (size_t)CDIM * CDIM);
  cvt_f32_to_f16<<<96, 256, 0, stream>>>(wmat, wh, (size_t)HEADS * MFEAT * HDIM);

  // 1) kqv = x @ kqv_w^T + kqv_b  -> f16 [32768 x 2304]
  gemm_wmma<_Float16><<<dim3(ROWS / 128, C3 / 64), 256, 0, stream>>>(
      xh, wkqvh, kqv_b, kqvh, ROWS, CDIM, C3);

  // 2) kp = prm_exp(k), qp = prm_exp(q)
  prm_exp_kernel<<<dim3(NSEQ / 128, BDIM * HEADS), 256, 0, stream>>>(kqvh, wh, kp, 0);
  prm_exp_kernel<<<dim3(NSEQ / 128, BDIM * HEADS), 256, 0, stream>>>(kqvh, wh, qp, 1);

  // 3) kptvT[m][d] = sum_t kp*v ; ks[m] = sum_t kp
  kptv_kernel<<<BDIM * HEADS, 256, 0, stream>>>(kqvh, kp, kptvT);
  ks_kernel<<<BDIM * HEADS, 256, 0, stream>>>(kp, ksbuf);

  // 4) y = (qp @ kptvT) / (qp . ks + eps)  -> f16 [32768 x 768]
  y_kernel<<<dim3(NSEQ / 128, BDIM * HEADS), 256, 0, stream>>>(qp, kptvT, ksbuf, yh);

  // 5) out = y @ proj_w^T + proj_b  -> f32
  gemm_wmma<float><<<dim3(ROWS / 128, CDIM / 64), 256, 0, stream>>>(
      yh, projh, proj_b, out, ROWS, CDIM, CDIM);
}
